// RNN_65738769433469
// MI455X (gfx1250) — 16-bit and 8-bit matrix operand order repaired and checked by reading, compiled, not run on hardware
//
#include <hip/hip_runtime.h>
#include <stdint.h>

typedef _Float16 v8h  __attribute__((ext_vector_type(8)));
typedef _Float16 v16h __attribute__((ext_vector_type(16)));
typedef float    v8f  __attribute__((ext_vector_type(8)));
typedef float    v4f  __attribute__((ext_vector_type(4)));
typedef int      v16i __attribute__((ext_vector_type(16)));

#define X_LDF 136
#define H_LD  72

__device__ __forceinline__ v16h cat16(v8h lo, v8h hi) {
  v16h r;
#pragma unroll
  for (int e = 0; e < 8; ++e) { r[e] = lo[e]; r[e + 8] = hi[e]; }
  return r;
}

__device__ __forceinline__ v16h lds_bfrag(const _Float16* S, int ld, int n, int kc, int half) {
  const _Float16* p = S + n * ld + kc * 32 + half * 8;
  return cat16(*(const v8h*)p, *(const v8h*)(p + 16));
}

__device__ __forceinline__ float sig_f(float x) {
  return __builtin_amdgcn_rcpf(1.0f + __expf(-x));
}
__device__ __forceinline__ float tanh_f(float x) {
  return 1.0f - 2.0f * __builtin_amdgcn_rcpf(1.0f + __expf(2.0f * x));
}

__device__ __forceinline__ uint32_t sw_fp8_1(float f) {
  _Float16 hv = (_Float16)f;
  uint16_t u = __builtin_bit_cast(uint16_t, hv);
  uint32_t sgn = (u >> 8) & 0x80u;
  int ex = (u >> 10) & 0x1f;
  uint32_t man = u & 0x3ffu;
  if (ex == 0) return sgn;
  if (ex == 31) return sgn | 0x7eu;
  int e8 = ex - 15 + 7;
  uint32_t m3 = (man + 0x40u) >> 7;
  if (m3 & 0x8u) { m3 = 0; ++e8; }
  if (e8 <= 0) return sgn;
  if (e8 > 15 || (e8 == 15 && m3 == 7)) { e8 = 15; m3 = 6; }
  return sgn | ((uint32_t)e8 << 3) | m3;
}
__device__ __forceinline__ int pk4_fp8(float a, float b, float c, float d) {
#if __has_builtin(__builtin_amdgcn_cvt_pk_fp8_f32)
  int v = 0;
  v = __builtin_amdgcn_cvt_pk_fp8_f32(a, b, v, false);
  v = __builtin_amdgcn_cvt_pk_fp8_f32(c, d, v, true);
  return v;
#else
  return (int)(sw_fp8_1(a) | (sw_fp8_1(b) << 8) | (sw_fp8_1(c) << 16) | (sw_fp8_1(d) << 24));
#endif
}

__device__ __forceinline__ void async_b128(const float* g, const float* l) {
  uint32_t loff = (uint32_t)(uintptr_t)l;
  asm volatile("global_load_async_to_lds_b128 %0, %1, off"
               :: "v"(loff), "v"(g) : "memory");
}
__device__ __forceinline__ void wait_async0() {
  asm volatile("s_wait_asynccnt 0x0" ::: "memory");
}

__global__ __launch_bounds__(128, 1)
void lstm_wmma_gfx1250(const float* __restrict__ x, const float* __restrict__ W_ih,
                       const float* __restrict__ W_hh, const float* __restrict__ b_ih,
                       const float* __restrict__ b_hh, const float* __restrict__ W_o,
                       const float* __restrict__ b_o, float* __restrict__ out)
{
  __shared__ __align__(16) float    sXf[2][16 * X_LDF];
  __shared__ __align__(16) _Float16 sH[2][16 * H_LD];
  __shared__ __align__(32) float    sBias[256];
  __shared__ __align__(32) float    sLogit[16 * 128];

  const int tid  = threadIdx.x;
  const int lane = tid & 31;
  const int wv   = tid >> 5;
  const int nn   = lane & 15;
  const int hf   = lane >> 4;

  v16i aW8[4];
#pragma unroll
  for (int gi = 0; gi < 4; ++gi) {
    const int gf = wv + 4 * gi;
    const int m  = 16 * gf + nn;
#pragma unroll
    for (int d = 0; d < 16; ++d) {
      const int K = (d >> 3) * 64 + ((d & 7) >> 1) * 16 + hf * 8 + (d & 1) * 4;
      const float* p = W_ih + (size_t)m * 128 + K;
      v4f q = *(const v4f*)p;
      aW8[gi][d] = pk4_fp8(q[0], q[1], q[2], q[3]);
    }
  }
  v16h aWhh[4][2];
#pragma unroll
  for (int gi = 0; gi < 4; ++gi) {
    const int m = 16 * (wv + 4 * gi) + nn;
#pragma unroll
    for (int kc = 0; kc < 2; ++kc) {
      const float* p = W_hh + (size_t)m * 64 + kc * 32 + hf * 8;
      v16h a;
#pragma unroll
      for (int e = 0; e < 8; ++e) { a[e] = (_Float16)p[e]; a[e + 8] = (_Float16)p[e + 16]; }
      aWhh[gi][kc] = a;
    }
  }

  for (int i = tid; i < 256; i += 128) sBias[i] = b_ih[i] + b_hh[i];
  for (int i = tid; i < 16 * H_LD; i += 128) { sH[0][i] = (_Float16)0.f; sH[1][i] = (_Float16)0.f; }

  v8f c;
#pragma unroll
  for (int r = 0; r < 8; ++r) c[r] = 0.f;

  const float* xblk = x + (size_t)blockIdx.x * 16 * 256 * 128;
  int crow[4], ccol[4];
#pragma unroll
  for (int k = 0; k < 4; ++k) { int cc = tid + 128 * k; crow[k] = cc >> 5; ccol[k] = (cc & 31) * 4; }

#pragma unroll
  for (int k = 0; k < 4; ++k)
    async_b128(xblk + (size_t)crow[k] * 256 * 128 + ccol[k],
               &sXf[0][crow[k] * X_LDF + ccol[k]]);
  wait_async0();
  __syncthreads();

#pragma unroll 1
  for (int t = 0; t < 256; ++t) {
    if (t < 255) {
#pragma unroll
      for (int k = 0; k < 4; ++k)
        async_b128(xblk + (size_t)crow[k] * 256 * 128 + (size_t)(t + 1) * 128 + ccol[k],
                   &sXf[(t + 1) & 1][crow[k] * X_LDF + ccol[k]]);
      __builtin_prefetch(xblk + (size_t)crow[0] * 256 * 128 + (size_t)(t + 2) * 128 + ccol[0], 0, 2);
    }

    const float* xs = &sXf[t & 1][nn * X_LDF];
    v16i bx;
#pragma unroll
    for (int g = 0; g < 4; ++g) {
      const float* p = xs + g * 32 + hf * 8;
      v4f q0 = *(const v4f*)(p + 0),  q1 = *(const v4f*)(p + 4);
      v4f q2 = *(const v4f*)(p + 16),  q3 = *(const v4f*)(p + 20);
      bx[g * 4 + 0] = pk4_fp8(q0[0], q0[1], q0[2], q0[3]);
      bx[g * 4 + 1] = pk4_fp8(q1[0], q1[1], q1[2], q1[3]);
      bx[g * 4 + 2] = pk4_fp8(q2[0], q2[1], q2[2], q2[3]);
      bx[g * 4 + 3] = pk4_fp8(q3[0], q3[1], q3[2], q3[3]);
    }
    v16h bh[2];
#pragma unroll
    for (int kc = 0; kc < 2; ++kc) bh[kc] = lds_bfrag(sH[t & 1], H_LD, nn, kc, hf);

    v8f acc[4];
#pragma unroll
    for (int gi = 0; gi < 4; ++gi) {
      const int gf = wv + 4 * gi;
      acc[gi] = *(const v8f*)&sBias[16 * gf + 8 * hf];
      acc[gi] = __builtin_amdgcn_wmma_f32_16x16x128_fp8_fp8(aW8[gi], bx,
                                                            (short)0, acc[gi], false, false);
#pragma unroll
      for (int kc = 0; kc < 2; ++kc)
        acc[gi] = __builtin_amdgcn_wmma_f32_16x16x32_f16(false, aWhh[gi][kc], false, bh[kc],
                                                         (short)0, acc[gi], false, false);
    }

    v8h hp;
#pragma unroll
    for (int r = 0; r < 8; ++r) {
      float iv = sig_f(acc[0][r]);
      float fv = sig_f(acc[1][r]);
      float gv = tanh_f(acc[2][r]);
      float ov = sig_f(acc[3][r]);
      float cn = fv * c[r] + iv * gv;
      c[r] = cn;
      hp[r] = (_Float16)(ov * tanh_f(cn));
    }
    *(v8h*)&sH[(t + 1) & 1][nn * H_LD + 16 * wv + 8 * hf] = hp;

    wait_async0();
    __syncthreads();
  }

  v16h rb[2];
#pragma unroll
  for (int kc = 0; kc < 2; ++kc) {
    v16h v = lds_bfrag(sH[0], H_LD, nn, kc, hf);
#pragma unroll
    for (int e = 0; e < 16; ++e) v[e] = (v[e] > (_Float16)0.f) ? v[e] : (_Float16)0.f;
    rb[kc] = v;
  }
#pragma unroll
  for (int p2 = 0; p2 < 2; ++p2) {
    const int of = 2 * wv + p2;
    v8f acc;
    {
      const float* bp = b_o + 16 * of + 8 * hf;
#pragma unroll
      for (int e = 0; e < 8; ++e) acc[e] = bp[e];
    }
#pragma unroll
    for (int kc = 0; kc < 2; ++kc) {
      const float* s = W_o + (size_t)(16 * of + nn) * 64 + kc * 32 + hf * 8;
      v16h a;
#pragma unroll
      for (int e = 0; e < 8; ++e) { a[e] = (_Float16)s[e]; a[e + 8] = (_Float16)s[e + 16]; }
      acc = __builtin_amdgcn_wmma_f32_16x16x32_f16(false, a, false, rb[kc],
                                                   (short)0, acc, false, false);
    }
    *(v8f*)&sLogit[nn * 128 + 16 * of + 8 * hf] = acc;
  }
  __syncthreads();

  if (tid < 16) {
    const float* row = &sLogit[tid * 128];
    float m = -3.4e38f;
    for (int o = 0; o < 128; ++o) m = fmaxf(m, row[o]);
    float ssum = 0.f;
    for (int o = 0; o < 128; ++o) ssum += __expf(row[o] - m);
    float ls = m + __logf(ssum);
    float* dst = out + ((size_t)blockIdx.x * 16 + tid) * 128;
    for (int o = 0; o < 128; ++o) dst[o] = row[o] - ls;
  }
}

extern "C" void kernel_launch(void* const* d_in, const int* in_sizes, int n_in,
                              void* d_out, int out_size, void* d_ws, size_t ws_size,
                              hipStream_t stream) {
  const float* x    = (const float*)d_in[0];
  const float* W_ih = (const float*)d_in[1];
  const float* W_hh = (const float*)d_in[2];
  const float* b_ih = (const float*)d_in[3];
  const float* b_hh = (const float*)d_in[4];
  const float* W_o  = (const float*)d_in[5];
  const float* b_o  = (const float*)d_in[6];
  float* out = (float*)d_out;

  dim3 grid(64);
  dim3 block(128);
  hipLaunchKernelGGL(lstm_wmma_gfx1250, grid, block, 0, stream,
                     x, W_ih, W_hh, b_ih, b_hh, W_o, b_o, out);
}
